// AdaLNDiTBlock_38706245271946
// MI455X (gfx1250) — matrix operand order repaired and checked by reading, not compiled, not run on hardware
//
#include <hip/hip_runtime.h>

// ---------------------------------------------------------------------------
// AdaLN DiT block for MI455X (gfx1250, wave32, WMMA bf16 -> fp32 accum)
// B=4, S=1024, D=1024, H=16, DK=64, HID=4096
// ---------------------------------------------------------------------------

#define B_    4
#define S_    1024
#define D_    1024
#define H_    16
#define DK_   64
#define HID_  4096
#define SIXD_ 6144

typedef __bf16 bf16;
typedef __attribute__((ext_vector_type(16))) __bf16 v16bf;
typedef __attribute__((ext_vector_type(8)))  __bf16 v8bf;
typedef __attribute__((ext_vector_type(8)))  float  v8f;

// D = A(16x32 bf16) x B(32x16 bf16) + C(16x16 f32)
static __device__ __forceinline__ v8f wmma_bf16(v16bf a, v16bf b, v8f c) {
  return __builtin_amdgcn_wmma_f32_16x16x32_bf16(false, a, false, b,
                                                 (short)0, c, false, false);
}

// Build a 16-element bf16 fragment from two contiguous 8-element chunks.
static __device__ __forceinline__ v16bf pack2(const bf16* p0, const bf16* p1) {
  v8bf a = *(const v8bf*)p0;
  v8bf b = *(const v8bf*)p1;
  v16bf r;
#pragma unroll
  for (int i = 0; i < 8; ++i) { r[i] = a[i]; r[i + 8] = b[i]; }
  return r;
}

// ---------------------------------------------------------------------------
// 1) SiLU on cond  (B*D elements)
// ---------------------------------------------------------------------------
__global__ void silu_kernel(const float* __restrict__ c, float* __restrict__ o,
                            int n) {
  int i = blockIdx.x * blockDim.x + threadIdx.x;
  if (i < n) { float v = c[i]; o[i] = v / (1.f + __expf(-v)); }
}

// ---------------------------------------------------------------------------
// 2) mod = silu(cond) @ Wada^T + bada   -> [B, 6D] fp32 (tiny GEMV)
// ---------------------------------------------------------------------------
__global__ __launch_bounds__(128) void ada_kernel(
    const float* __restrict__ sc, const float* __restrict__ Wada,
    const float* __restrict__ bada, float* __restrict__ mod) {
  int idx = blockIdx.x * blockDim.x + threadIdx.x;
  if (idx >= B_ * SIXD_) return;
  int b = idx / SIXD_, o = idx - b * SIXD_;
  const float4* cp = (const float4*)(sc + (size_t)b * D_);
  const float4* wp = (const float4*)(Wada + (size_t)o * D_);
  float acc = 0.f;
#pragma unroll 4
  for (int i = 0; i < D_ / 4; ++i) {
    float4 a = cp[i], w = wp[i];
    acc += a.x * w.x + a.y * w.y + a.z * w.z + a.w * w.w;
  }
  mod[idx] = acc + bada[o];
}

// ---------------------------------------------------------------------------
// 3) fp32 -> bf16 weight conversion
// ---------------------------------------------------------------------------
__global__ void cvt_bf16_kernel(const float* __restrict__ in,
                                bf16* __restrict__ o, int n) {
  int i = blockIdx.x * blockDim.x + threadIdx.x;
  if (i < n) o[i] = (bf16)in[i];
}

// ---------------------------------------------------------------------------
// 4) LayerNorm (eps=1e-6, no affine) + adaLN modulate -> bf16
//    out = ln(x) * (1 + mod[scale_idx]) + mod[shift_idx]
//    One block per row of 1024; 256 threads, 4 elements each.
// ---------------------------------------------------------------------------
__global__ __launch_bounds__(256) void ln_mod_kernel(
    const float* __restrict__ x, const float* __restrict__ mod,
    bf16* __restrict__ out, int shift_idx, int scale_idx) {
  __shared__ float r1[256];
  __shared__ float r2[256];
  const int tid = threadIdx.x;
  const int row = blockIdx.x;
  const int b = row >> 10;  // row / S
  const float4 xv = ((const float4*)(x + (size_t)row * D_))[tid];
  float s  = xv.x + xv.y + xv.z + xv.w;
  float sq = xv.x * xv.x + xv.y * xv.y + xv.z * xv.z + xv.w * xv.w;
  r1[tid] = s; r2[tid] = sq;
  __syncthreads();
#pragma unroll
  for (int st = 128; st > 0; st >>= 1) {
    if (tid < st) { r1[tid] += r1[tid + st]; r2[tid] += r2[tid + st]; }
    __syncthreads();
  }
  const float mu   = r1[0] * (1.f / D_);
  const float var  = r2[0] * (1.f / D_) - mu * mu;
  const float rstd = rsqrtf(var + 1e-6f);
  const float4 sc = ((const float4*)(mod + (size_t)b * SIXD_ + scale_idx * D_))[tid];
  const float4 sh = ((const float4*)(mod + (size_t)b * SIXD_ + shift_idx * D_))[tid];
  bf16* op = out + (size_t)row * D_ + tid * 4;
  op[0] = (bf16)((xv.x - mu) * rstd * (1.f + sc.x) + sh.x);
  op[1] = (bf16)((xv.y - mu) * rstd * (1.f + sc.y) + sh.y);
  op[2] = (bf16)((xv.z - mu) * rstd * (1.f + sc.z) + sh.z);
  op[3] = (bf16)((xv.w - mu) * rstd * (1.f + sc.w) + sh.w);
}

// ---------------------------------------------------------------------------
// 5) Generic WMMA GEMM:  C[M,N] = A[M,K](bf16) @ W[N,K]^T(bf16) + bias
//    Block tile 128x256 (8 waves: 2 along M x 4 along N).
//    Wave tile 64x64 = 4x4 wmma tiles -> 16 wmma per 16 b128 loads per k-step.
//    Epilogues selected at compile time.
// ---------------------------------------------------------------------------
enum { EPI_BF16 = 0, EPI_VT = 1, EPI_RESID = 2, EPI_GELU = 3 };

template <int MODE>
__global__ __launch_bounds__(256) void gemm_wmma(
    const bf16* __restrict__ A, const bf16* __restrict__ W,
    const float* __restrict__ bias, const float* __restrict__ resid,
    const float* __restrict__ mod, int alpha_idx,
    bf16* __restrict__ outb, float* __restrict__ outf,
    int M, int N, int K) {
  const int lane = threadIdx.x & 31;
  const int wid  = threadIdx.x >> 5;
  const int n15  = lane & 15;
  const int hi   = lane >> 4;
  const int wm   = wid & 1;   // 2 waves along M
  const int wn   = wid >> 1;  // 4 waves along N
  const int m0   = blockIdx.x * 128 + wm * 64;
  const int n0   = blockIdx.y * 256 + wn * 64;

  v8f acc[4][4];
#pragma unroll
  for (int i = 0; i < 4; ++i)
#pragma unroll
    for (int j = 0; j < 4; ++j)
#pragma unroll
      for (int r = 0; r < 8; ++r) acc[i][j][r] = 0.f;

  // A-fragment: lane holds row (lane&15); 16 contiguous K at 16*hi.
  const bf16* ap = A + (size_t)(m0 + n15) * K + 16 * hi;
  // B-fragment: lane holds W row (n-col); 16 contiguous K at 16*hi.
  const bf16* wp = W + (size_t)(n0 + n15) * K + 16 * hi;
  const size_t rowK16 = (size_t)16 * K;

  for (int k0 = 0; k0 < K; k0 += 32) {
    __builtin_prefetch(ap + k0 + 256, 0, 0);  // global_prefetch next A chunk
    v16bf a[4];
#pragma unroll
    for (int mt = 0; mt < 4; ++mt)
      a[mt] = pack2(ap + (size_t)mt * rowK16 + k0,
                    ap + (size_t)mt * rowK16 + k0 + 8);
#pragma unroll
    for (int nt = 0; nt < 4; ++nt) {
      v16bf bb = *(const v16bf*)(wp + (size_t)nt * rowK16 + k0);
      acc[0][nt] = wmma_bf16(a[0], bb, acc[0][nt]);
      acc[1][nt] = wmma_bf16(a[1], bb, acc[1][nt]);
      acc[2][nt] = wmma_bf16(a[2], bb, acc[2][nt]);
      acc[3][nt] = wmma_bf16(a[3], bb, acc[3][nt]);
    }
  }

  // Epilogue: C layout -> row = base + r + 8*hi, col = base + (lane&15)
#pragma unroll
  for (int mt = 0; mt < 4; ++mt) {
#pragma unroll
    for (int nt = 0; nt < 4; ++nt) {
      const int col = n0 + nt * 16 + n15;
      const float bv = bias[col];
#pragma unroll
      for (int r = 0; r < 8; ++r) {
        const int row = m0 + mt * 16 + r + 8 * hi;
        float v = acc[mt][nt][r] + bv;
        if (MODE == EPI_BF16) {
          outb[(size_t)row * N + col] = (bf16)v;
        } else if (MODE == EPI_VT) {
          // V transposed per head: vt[b][h][dk][s]
          const int bb = row >> 10, ss = row & (S_ - 1);
          const int hd = col >> 6,  dk = col & (DK_ - 1);
          outb[(((size_t)bb * H_ + hd) * DK_ + dk) * S_ + ss] = (bf16)v;
        } else if (MODE == EPI_GELU) {
          const float g = 0.5f * v * (1.f + erff(v * 0.70710678118654752f));
          outb[(size_t)row * N + col] = (bf16)g;
        } else {  // EPI_RESID: out = resid + alpha * (gemm + bias), fp32
          const int bb = row >> 10;
          const float al = mod[(size_t)bb * SIXD_ + alpha_idx * D_ + col];
          outf[(size_t)row * N + col] =
              resid[(size_t)row * N + col] + al * v;
        }
      }
    }
  }
}

// ---------------------------------------------------------------------------
// 6) Flash attention. One block = (b, h, 64 q-rows); 4 waves, each wave owns
//    a 16-row q tile with full Q fragments resident (dk=64 -> two A frags).
//    Streams K/V in 32-token chunks; online softmax in fp32; P tile staged
//    through per-wave LDS to re-lay it as an A fragment for P@V.
// ---------------------------------------------------------------------------
__global__ __launch_bounds__(128) void attn_kernel(
    const bf16* __restrict__ q, const bf16* __restrict__ k,
    const bf16* __restrict__ vt, bf16* __restrict__ out) {
  __shared__ __align__(16) bf16 pbuf[4][16][32];
  const int lane = threadIdx.x & 31;
  const int wid  = threadIdx.x >> 5;
  const int n15  = lane & 15;
  const int hi   = lane >> 4;
  const int blk = blockIdx.x;
  const int qb = blk & 15;   // S/64 = 16 q-blocks
  const int bh = blk >> 4;
  const int h  = bh & (H_ - 1);
  const int b  = bh >> 4;
  const int q0 = qb * 64 + wid * 16;

  // Q fragments: dk 0..31 and dk 32..63
  const bf16* qr = q + ((size_t)(b * S_ + q0 + n15) * D_ + h * DK_);
  const v16bf aq0 = pack2(qr + 16 * hi,      qr + 8 + 16 * hi);
  const v16bf aq1 = pack2(qr + 32 + 16 * hi, qr + 40 + 16 * hi);

  v8f o0, o1, o2, o3;
  float mrow[8], lrow[8];
#pragma unroll
  for (int r = 0; r < 8; ++r) {
    o0[r] = o1[r] = o2[r] = o3[r] = 0.f;
    mrow[r] = -3.0e38f;
    lrow[r] = 0.f;
  }

  const bf16* kbase = k + ((size_t)b * S_ * D_ + h * DK_ + 16 * hi);
  const bf16* vbase = vt + (((size_t)(b * H_ + h) * DK_ + n15) * S_ + 16 * hi);

  for (int c0 = 0; c0 < S_; c0 += 32) {
    // scores: two 16x16 tiles (columns c0..c0+15, c0+16..c0+31)
    const bf16* kr0 = kbase + (size_t)(c0 + n15) * D_;
    const bf16* kr1 = kr0 + (size_t)16 * D_;
    v8f s0, s1;
#pragma unroll
    for (int r = 0; r < 8; ++r) { s0[r] = 0.f; s1[r] = 0.f; }
    s0 = wmma_bf16(aq0, *(const v16bf*)kr0, s0);
    s0 = wmma_bf16(aq1, *(const v16bf*)(kr0 + 32), s0);
    s1 = wmma_bf16(aq0, *(const v16bf*)kr1, s1);
    s1 = wmma_bf16(aq1, *(const v16bf*)(kr1 + 32), s1);

    // online softmax over the 32 new columns (scale = 1/sqrt(64) = 0.125)
#pragma unroll
    for (int r = 0; r < 8; ++r) {
      float a = s0[r] * 0.125f;
      float c = s1[r] * 0.125f;
      float mx = fmaxf(a, c);
#pragma unroll
      for (int off = 1; off < 16; off <<= 1)
        mx = fmaxf(mx, __shfl_xor(mx, off, 16));
      const float mn  = fmaxf(mrow[r], mx);
      const float fac = __expf(mrow[r] - mn);
      mrow[r] = mn;
      const float p0 = __expf(a - mn);
      const float p1 = __expf(c - mn);
      float sm = p0 + p1;
#pragma unroll
      for (int off = 1; off < 16; off <<= 1) sm += __shfl_xor(sm, off, 16);
      lrow[r] = lrow[r] * fac + sm;
      pbuf[wid][r + 8 * hi][n15]      = (bf16)p0;
      pbuf[wid][r + 8 * hi][16 + n15] = (bf16)p1;
      o0[r] *= fac; o1[r] *= fac; o2[r] *= fac; o3[r] *= fac;
    }
    // Same-wave DS ops are in-order; fence compiler + hw counter explicitly.
    asm volatile("s_wait_dscnt 0" ::: "memory");

    // Re-lay P (16 rows x 32 tokens) as WMMA A-fragment.
    const v16bf pa =
        pack2(&pbuf[wid][n15][16 * hi], &pbuf[wid][n15][8 + 16 * hi]);

    // O += P @ V  (V transposed: contiguous along tokens)
    const bf16* vr = vbase + c0;
    o0 = wmma_bf16(pa, *(const v16bf*)vr, o0);
    o1 = wmma_bf16(pa, *(const v16bf*)(vr + (size_t)16 * S_), o1);
    o2 = wmma_bf16(pa, *(const v16bf*)(vr + (size_t)32 * S_), o2);
    o3 = wmma_bf16(pa, *(const v16bf*)(vr + (size_t)48 * S_), o3);
  }

#pragma unroll
  for (int r = 0; r < 8; ++r) {
    const float inv = 1.f / lrow[r];
    const size_t base =
        (size_t)(b * S_ + q0 + r + 8 * hi) * D_ + h * DK_ + n15;
    out[base]      = (bf16)(o0[r] * inv);
    out[base + 16] = (bf16)(o1[r] * inv);
    out[base + 32] = (bf16)(o2[r] * inv);
    out[base + 48] = (bf16)(o3[r] * inv);
  }
}

// ---------------------------------------------------------------------------
// Launch sequence
// ---------------------------------------------------------------------------
extern "C" void kernel_launch(void* const* d_in, const int* in_sizes, int n_in,
                              void* d_out, int out_size, void* d_ws,
                              size_t ws_size, hipStream_t stream) {
  const float* x    = (const float*)d_in[0];
  const float* cond = (const float*)d_in[1];
  const float* Wq   = (const float*)d_in[2];
  const float* bq   = (const float*)d_in[3];
  const float* Wk   = (const float*)d_in[4];
  const float* bk   = (const float*)d_in[5];
  const float* Wv   = (const float*)d_in[6];
  const float* bv   = (const float*)d_in[7];
  const float* Wo   = (const float*)d_in[8];
  const float* bo   = (const float*)d_in[9];
  const float* W1   = (const float*)d_in[10];
  const float* b1   = (const float*)d_in[11];
  const float* W2   = (const float*)d_in[12];
  const float* b2   = (const float*)d_in[13];
  const float* Wada = (const float*)d_in[14];
  const float* bada = (const float*)d_in[15];

  char* ws = (char*)d_ws;
  const size_t MB = (size_t)1 << 20;
  float* silu  = (float*)(ws);                   // 16 KB
  float* mod   = (float*)(ws + 64 * 1024);       // 96 KB
  bf16*  hbuf  = (bf16*)(ws + 1 * MB);           // 8 MB  (h / attn / h2)
  bf16*  qbuf  = (bf16*)(ws + 9 * MB);           // 8 MB
  bf16*  kbuf  = (bf16*)(ws + 17 * MB);          // 8 MB
  bf16*  vtbuf = (bf16*)(ws + 25 * MB);          // 8 MB
  float* x2    = (float*)(ws + 33 * MB);         // 16 MB
  bf16*  hid   = (bf16*)(ws + 49 * MB);          // 32 MB
  bf16*  wq16  = (bf16*)(ws + 81 * MB);          // 2 MB
  bf16*  wk16  = (bf16*)(ws + 83 * MB);          // 2 MB
  bf16*  wv16  = (bf16*)(ws + 85 * MB);          // 2 MB
  bf16*  wo16  = (bf16*)(ws + 87 * MB);          // 2 MB
  bf16*  w116  = (bf16*)(ws + 89 * MB);          // 8 MB
  bf16*  w216  = (bf16*)(ws + 97 * MB);          // 8 MB (total 105 MB)

  // adaLN modulation
  silu_kernel<<<(B_ * D_ + 255) / 256, 256, 0, stream>>>(cond, silu, B_ * D_);
  ada_kernel<<<(B_ * SIXD_ + 127) / 128, 128, 0, stream>>>(silu, Wada, bada,
                                                           mod);
  // weight conversions (fp32 -> bf16)
  cvt_bf16_kernel<<<(D_ * D_ + 255) / 256, 256, 0, stream>>>(Wq, wq16, D_ * D_);
  cvt_bf16_kernel<<<(D_ * D_ + 255) / 256, 256, 0, stream>>>(Wk, wk16, D_ * D_);
  cvt_bf16_kernel<<<(D_ * D_ + 255) / 256, 256, 0, stream>>>(Wv, wv16, D_ * D_);
  cvt_bf16_kernel<<<(D_ * D_ + 255) / 256, 256, 0, stream>>>(Wo, wo16, D_ * D_);
  cvt_bf16_kernel<<<(HID_ * D_ + 255) / 256, 256, 0, stream>>>(W1, w116,
                                                               HID_ * D_);
  cvt_bf16_kernel<<<(D_ * HID_ + 255) / 256, 256, 0, stream>>>(W2, w216,
                                                               D_ * HID_);

  // attention branch
  ln_mod_kernel<<<B_ * S_, 256, 0, stream>>>(x, mod, hbuf, /*shift=*/0,
                                             /*scale=*/1);
  dim3 blk(256);
  dim3 gD(32, 4);  // M=4096/128, N=1024/256
  gemm_wmma<EPI_BF16><<<gD, blk, 0, stream>>>(hbuf, wq16, bq, nullptr, nullptr,
                                              0, qbuf, nullptr, 4096, 1024,
                                              1024);
  gemm_wmma<EPI_BF16><<<gD, blk, 0, stream>>>(hbuf, wk16, bk, nullptr, nullptr,
                                              0, kbuf, nullptr, 4096, 1024,
                                              1024);
  gemm_wmma<EPI_VT><<<gD, blk, 0, stream>>>(hbuf, wv16, bv, nullptr, nullptr,
                                            0, vtbuf, nullptr, 4096, 1024,
                                            1024);
  attn_kernel<<<B_ * H_ * (S_ / 64), 128, 0, stream>>>(qbuf, kbuf, vtbuf,
                                                       hbuf);
  gemm_wmma<EPI_RESID><<<gD, blk, 0, stream>>>(hbuf, wo16, bo, x, mod,
                                               /*alpha=*/2, nullptr, x2, 4096,
                                               1024, 1024);

  // MLP branch
  ln_mod_kernel<<<B_ * S_, 256, 0, stream>>>(x2, mod, hbuf, /*shift=*/3,
                                             /*scale=*/4);
  dim3 gH(32, 16);  // N=4096/256
  gemm_wmma<EPI_GELU><<<gH, blk, 0, stream>>>(hbuf, w116, b1, nullptr, nullptr,
                                              0, hid, nullptr, 4096, 4096,
                                              1024);
  gemm_wmma<EPI_RESID><<<gD, blk, 0, stream>>>(hid, w216, b2, x2, mod,
                                               /*alpha=*/5, nullptr,
                                               (float*)d_out, 4096, 1024,
                                               4096);
}
